// NonLocal_82128364634387
// MI455X (gfx1250) — compile-verified, matrix operand order checked by reading, not run on hardware
//
#include <hip/hip_runtime.h>
#include <hip/hip_bf16.h>

typedef __bf16 bf16_t;
typedef __attribute__((ext_vector_type(16))) __bf16 v16bf;
typedef __attribute__((ext_vector_type(8)))  __bf16 v8bf;
typedef __attribute__((ext_vector_type(8)))  float  v8f;

#define NB    8      // batch
#define CIN   256
#define CHD   128
#define NSP   3136   // 56*56 ; 3136 = 49 * 64
#define NTILE 49
#define L2E   1.44269504088896340736f
#define QSCALE 0.08838834764831845f   // 1/sqrt(128)

// ---- fragment helpers -------------------------------------------------------

__device__ __forceinline__ v16bf ld_frag(const bf16_t* lo, const bf16_t* hi) {
    v8bf a = *reinterpret_cast<const v8bf*>(lo);
    v8bf b = *reinterpret_cast<const v8bf*>(hi);
    return __builtin_shufflevector(a, b, 0,1,2,3,4,5,6,7,8,9,10,11,12,13,14,15);
}

__device__ __forceinline__ v8f wmma_bf16(v16bf a, v16bf b, v8f c) {
    return __builtin_amdgcn_wmma_f32_16x16x32_bf16(
        /*neg_a=*/false, a, /*neg_b=*/false, b,
        /*c_mod=*/(short)0, c, /*reuse_a=*/false, /*reuse_b=*/false);
}

// ---- kernel 1: QKV projection (Wx + b), bf16 output (B, N, 128) -------------
// grid (49, 3, 8), block 256 (8 waves). Each wave: 16 rows x 64 cols.

__global__ __launch_bounds__(256)
void qkv_proj_kernel(const float* __restrict__ x,
                     const float* __restrict__ wq, const float* __restrict__ bq,
                     const float* __restrict__ wk, const float* __restrict__ bk,
                     const float* __restrict__ wv, const float* __restrict__ bv,
                     bf16_t* __restrict__ Qw, bf16_t* __restrict__ Kw,
                     bf16_t* __restrict__ Vw)
{
    const int tile  = blockIdx.x;
    const int which = blockIdx.y;          // 0=q 1=k 2=v
    const int b     = blockIdx.z;
    const int n0    = tile * 64;

    const float* w    = which == 0 ? wq : (which == 1 ? wk : wv);
    const float* bias = which == 0 ? bq : (which == 1 ? bk : bv);
    bf16_t*      outp = which == 0 ? Qw : (which == 1 ? Kw : Vw);
    const float  scale = which == 0 ? QSCALE : 1.0f;

    __shared__ alignas(16) bf16_t As[64][32];    // x^T tile  (m, k)
    __shared__ alignas(16) bf16_t Bs[128][32];   // w   tile  (oc, k)

    const int tid  = threadIdx.x;
    const int lane = tid & 31;
    const int wid  = tid >> 5;            // 0..7
    const int rowblk  = wid >> 1;         // 0..3
    const int colbase = (wid & 1) * 64;
    const int lrow = lane & 15;
    const int half = lane >> 4;

    v8f acc[4];
#pragma unroll
    for (int t = 0; t < 4; ++t) {
        float b0 = bias[colbase + t * 16 + lrow];
#pragma unroll
        for (int j = 0; j < 8; ++j) acc[t][j] = b0;
    }

#pragma unroll 1
    for (int ks = 0; ks < 8; ++ks) {
        // As[m][c] = x[b][ks*32+c][n0+m] (global coalesced along m)
#pragma unroll
        for (int i = 0; i < 8; ++i) {
            int idx = tid + i * 256;          // 0..2047
            int c = idx >> 6, m = idx & 63;
            As[m][c] = (bf16_t)x[((size_t)b * CIN + ks * 32 + c) * NSP + n0 + m];
        }
        // Bs[oc][c] = w[oc][ks*32+c]
#pragma unroll
        for (int i = 0; i < 16; ++i) {
            int idx = tid + i * 256;          // 0..4095
            int oc = idx >> 5, c = idx & 31;
            Bs[oc][c] = (bf16_t)w[(size_t)oc * CIN + ks * 32 + c];
        }
        __syncthreads();

        v16bf af = ld_frag(&As[rowblk * 16 + lrow][half * 8],
                           &As[rowblk * 16 + lrow][16 + half * 8]);
#pragma unroll
        for (int t = 0; t < 4; ++t) {
            int oc = colbase + t * 16 + lrow;
            v16bf bf = ld_frag(&Bs[oc][half * 8], &Bs[oc][16 + half * 8]);
            acc[t] = wmma_bf16(af, bf, acc[t]);
        }
        __syncthreads();
    }

#pragma unroll
    for (int t = 0; t < 4; ++t) {
        int oc = colbase + t * 16 + lrow;
#pragma unroll
        for (int j = 0; j < 8; ++j) {
            int n = n0 + rowblk * 16 + half * 8 + j;
            outp[((size_t)b * NSP + n) * CHD + oc] = (bf16_t)(acc[t][j] * scale);
        }
    }
}

// ---- kernel 2: flash attention ---------------------------------------------
// grid (49, 8), block 128 (4 waves, 16 query rows each). Streams 49 K/V tiles.

__global__ __launch_bounds__(128)
void attn_kernel(const bf16_t* __restrict__ Qw, const bf16_t* __restrict__ Kw,
                 const bf16_t* __restrict__ Vw, bf16_t* __restrict__ Ow)
{
    const int tile = blockIdx.x;
    const int b    = blockIdx.y;
    const int n0   = tile * 64;
    const int tid  = threadIdx.x;
    const int lane = tid & 31;
    const int wid  = tid >> 5;            // 0..3
    const int lrow = lane & 15;
    const int half = lane >> 4;

    __shared__ alignas(16) bf16_t Ks[64][128];   // (key, c)
    __shared__ alignas(16) bf16_t VsT[128][64];  // (c, key)
    __shared__ alignas(16) bf16_t Ps[4][16][64]; // per-wave P scratch

    // Q fragments resident in registers (4 k-steps of 32)
    v16bf qf[4];
    {
        const bf16_t* qrow = Qw + ((size_t)b * NSP + n0 + wid * 16 + lrow) * CHD;
#pragma unroll
        for (int ks = 0; ks < 4; ++ks)
            qf[ks] = ld_frag(qrow + ks * 32 + half * 8,
                             qrow + ks * 32 + 16 + half * 8);
    }

    float mrow[8], lsum[8];
#pragma unroll
    for (int j = 0; j < 8; ++j) { mrow[j] = -__builtin_inff(); lsum[j] = 0.f; }
    v8f oacc[8];
#pragma unroll
    for (int t = 0; t < 8; ++t)
#pragma unroll
        for (int j = 0; j < 8; ++j) oacc[t][j] = 0.f;

#pragma unroll 1
    for (int kt = 0; kt < NTILE; ++kt) {
        const int k0 = kt * 64;

        // prefetch next K/V tiles into cache hierarchy
        if (kt + 1 < NTILE) {
            const char* pk = (const char*)(Kw + ((size_t)b * NSP + k0 + 64) * CHD);
            const char* pv = (const char*)(Vw + ((size_t)b * NSP + k0 + 64) * CHD);
            __builtin_prefetch(pk + tid * 128, 0, 1);
            __builtin_prefetch(pv + tid * 128, 0, 1);
        }

        // K tile: straight vectorized copy (64x128 bf16 = 1024 uint4)
        {
            const uint4* src = reinterpret_cast<const uint4*>(Kw + ((size_t)b * NSP + k0) * CHD);
            uint4* dst = reinterpret_cast<uint4*>(&Ks[0][0]);
#pragma unroll
            for (int i = 0; i < 8; ++i) dst[tid + i * 128] = src[tid + i * 128];
        }
        // V tile: transpose into VsT (vector read, scalar LDS writes)
        {
            const uint4* src = reinterpret_cast<const uint4*>(Vw + ((size_t)b * NSP + k0) * CHD);
#pragma unroll
            for (int i = 0; i < 8; ++i) {
                int v = tid + i * 128;            // 0..1023
                int key = v >> 4;                 // 16 vecs per key row
                int cb  = (v & 15) * 8;
                uint4 d = src[v];
                const bf16_t* e = reinterpret_cast<const bf16_t*>(&d);
#pragma unroll
                for (int q = 0; q < 8; ++q) VsT[cb + q][key] = e[q];
            }
        }
        __syncthreads();

        // S = Q * K^T   (4 key-subtiles x 4 k-steps)
        v8f s[4];
#pragma unroll
        for (int t = 0; t < 4; ++t)
#pragma unroll
            for (int j = 0; j < 8; ++j) s[t][j] = 0.f;
#pragma unroll
        for (int t = 0; t < 4; ++t) {
#pragma unroll
            for (int ks = 0; ks < 4; ++ks) {
                v16bf kf = ld_frag(&Ks[t * 16 + lrow][ks * 32 + half * 8],
                                   &Ks[t * 16 + lrow][ks * 32 + 16 + half * 8]);
                s[t] = wmma_bf16(qf[ks], kf, s[t]);
            }
        }

        // online softmax, per row j (row = half*8+j, values across 16 lanes)
#pragma unroll
        for (int j = 0; j < 8; ++j) {
            float rm = fmaxf(fmaxf(s[0][j], s[1][j]), fmaxf(s[2][j], s[3][j]));
#pragma unroll
            for (int off = 8; off >= 1; off >>= 1)
                rm = fmaxf(rm, __shfl_xor(rm, off, 32));
            float nm    = fmaxf(mrow[j], rm);
            float alpha = exp2f((mrow[j] - nm) * L2E);
            float ps = 0.f;
#pragma unroll
            for (int t = 0; t < 4; ++t) {
                float p = exp2f((s[t][j] - nm) * L2E);
                s[t][j] = p;
                ps += p;
            }
#pragma unroll
            for (int off = 8; off >= 1; off >>= 1)
                ps += __shfl_xor(ps, off, 32);
            lsum[j] = lsum[j] * alpha + ps;
            mrow[j] = nm;
#pragma unroll
            for (int t = 0; t < 8; ++t) oacc[t][j] *= alpha;
        }

        // P -> per-wave LDS (bf16), re-read as A fragments
#pragma unroll
        for (int t = 0; t < 4; ++t)
#pragma unroll
            for (int j = 0; j < 8; ++j)
                Ps[wid][half * 8 + j][t * 16 + lrow] = (bf16_t)s[t][j];

        // O += P * V   (2 k-steps of 32 keys x 8 channel-subtiles)
#pragma unroll
        for (int ks = 0; ks < 2; ++ks) {
            v16bf pf = ld_frag(&Ps[wid][lrow][ks * 32 + half * 8],
                               &Ps[wid][lrow][ks * 32 + 16 + half * 8]);
#pragma unroll
            for (int t = 0; t < 8; ++t) {
                v16bf vf = ld_frag(&VsT[t * 16 + lrow][ks * 32 + half * 8],
                                   &VsT[t * 16 + lrow][ks * 32 + 16 + half * 8]);
                oacc[t] = wmma_bf16(pf, vf, oacc[t]);
            }
        }
        __syncthreads();
    }

    // epilogue: O /= l, write bf16 (B, N, 128)
#pragma unroll
    for (int t = 0; t < 8; ++t) {
        int c = t * 16 + lrow;
#pragma unroll
        for (int j = 0; j < 8; ++j) {
            int n = n0 + wid * 16 + half * 8 + j;
            Ow[((size_t)b * NSP + n) * CHD + c] = (bf16_t)(oacc[t][j] / lsum[j]);
        }
    }
}

// ---- kernel 3: out-projection + bias + residual -----------------------------
// grid (49, 8), block 256 (8 waves). Each wave: 16 rows x 128 output chans.

__global__ __launch_bounds__(256)
void oproj_kernel(const bf16_t* __restrict__ Ow, const float* __restrict__ wo,
                  const float* __restrict__ bo, const float* __restrict__ x,
                  float* __restrict__ out)
{
    const int tile = blockIdx.x;
    const int b    = blockIdx.y;
    const int n0   = tile * 64;
    const int tid  = threadIdx.x;
    const int lane = tid & 31;
    const int wid  = tid >> 5;
    const int rowblk  = wid >> 1;
    const int colhalf = wid & 1;
    const int lrow = lane & 15;
    const int half = lane >> 4;

    __shared__ alignas(16) bf16_t Ws[256][32];   // wo slice (oc, k)

    v8f acc[8];
#pragma unroll
    for (int t = 0; t < 8; ++t) {
        float b0 = bo[colhalf * 128 + t * 16 + lrow];
#pragma unroll
        for (int j = 0; j < 8; ++j) acc[t][j] = b0;
    }

#pragma unroll 1
    for (int ks = 0; ks < 4; ++ks) {
#pragma unroll
        for (int i = 0; i < 32; ++i) {
            int idx = tid + i * 256;          // 0..8191
            int oc = idx >> 5, c = idx & 31;
            Ws[oc][c] = (bf16_t)wo[(size_t)oc * CHD + ks * 32 + c];
        }
        __syncthreads();

        const bf16_t* orow = Ow + ((size_t)b * NSP + n0 + rowblk * 16 + lrow) * CHD + ks * 32;
        v16bf af = ld_frag(orow + half * 8, orow + 16 + half * 8);
#pragma unroll
        for (int t = 0; t < 8; ++t) {
            int oc = colhalf * 128 + t * 16 + lrow;
            v16bf bfr = ld_frag(&Ws[oc][half * 8], &Ws[oc][16 + half * 8]);
            acc[t] = wmma_bf16(af, bfr, acc[t]);
        }
        __syncthreads();
    }

    // residual + store: rows j=0..7 are contiguous along n -> float4 x2
#pragma unroll
    for (int t = 0; t < 8; ++t) {
        int oc = colhalf * 128 + t * 16 + lrow;
        size_t base = ((size_t)b * CIN + oc) * NSP + n0 + rowblk * 16 + half * 8;
        float4 x0 = *reinterpret_cast<const float4*>(x + base);
        float4 x1 = *reinterpret_cast<const float4*>(x + base + 4);
        float4 r0, r1;
        r0.x = x0.x + acc[t][0]; r0.y = x0.y + acc[t][1];
        r0.z = x0.z + acc[t][2]; r0.w = x0.w + acc[t][3];
        r1.x = x1.x + acc[t][4]; r1.y = x1.y + acc[t][5];
        r1.z = x1.z + acc[t][6]; r1.w = x1.w + acc[t][7];
        *reinterpret_cast<float4*>(out + base)     = r0;
        *reinterpret_cast<float4*>(out + base + 4) = r1;
    }
}

// ---- launcher ---------------------------------------------------------------

extern "C" void kernel_launch(void* const* d_in, const int* in_sizes, int n_in,
                              void* d_out, int out_size, void* d_ws, size_t ws_size,
                              hipStream_t stream)
{
    const float* x  = (const float*)d_in[0];
    const float* wq = (const float*)d_in[1];
    const float* bq = (const float*)d_in[2];
    const float* wk = (const float*)d_in[3];
    const float* bk = (const float*)d_in[4];
    const float* wv = (const float*)d_in[5];
    const float* bv = (const float*)d_in[6];
    const float* wo = (const float*)d_in[7];
    const float* bo = (const float*)d_in[8];
    float* out = (float*)d_out;

    const size_t seg = (size_t)NB * NSP * CHD;   // elements per bf16 tensor
    bf16_t* Qw = (bf16_t*)d_ws;
    bf16_t* Kw = Qw + seg;
    bf16_t* Vw = Kw + seg;
    bf16_t* Ow = Vw + seg;

    dim3 g1(NTILE, 3, NB);
    qkv_proj_kernel<<<g1, 256, 0, stream>>>(x, wq, bq, wk, bk, wv, bv, Qw, Kw, Vw);

    dim3 g2(NTILE, NB);
    attn_kernel<<<g2, 128, 0, stream>>>(Qw, Kw, Vw, Ow);

    dim3 g3(NTILE, NB);
    oproj_kernel<<<g3, 256, 0, stream>>>(Ow, wo, bo, x, out);
}
